// NeuronMoonlightAttention_90469191123379
// MI455X (gfx1250) — compile-verified, matrix operand order checked by reading, not run on hardware
//
#include <hip/hip_runtime.h>

// ---------- types ----------
typedef __bf16 bf16_t;
typedef __attribute__((ext_vector_type(16))) __bf16 v16bf;
typedef __attribute__((ext_vector_type(8)))  float  v8f;
typedef __attribute__((ext_vector_type(4)))  unsigned int u32x4;
typedef __attribute__((ext_vector_type(4)))  unsigned int u32x4t;
typedef __attribute__((ext_vector_type(8)))  int i32x8t;
typedef __attribute__((ext_vector_type(4)))  int i32x4t;

union FragB { u32x4 u[2]; v16bf v; };

// ---------- problem constants ----------
#define BB   2
#define SS   2048
#define HH   2048
#define NHH  16
#define NOPE 128
#define ROPE 64
#define VDIM 128
#define RANK 512
#define QHD  192   // NOPE + ROPE
#define DTOT 576   // RANK + ROPE

// ---------- Tensor Data Mover availability ----------
#if defined(__has_builtin)
#if __has_builtin(__builtin_amdgcn_tensor_load_to_lds) && \
    __has_builtin(__builtin_amdgcn_s_wait_tensorcnt)
#define HAS_TDM 1
#endif
#endif
#ifndef HAS_TDM
#define HAS_TDM 0
#endif

#if HAS_TDM
// LDS byte offset = low 32 bits of the generic pointer (LDS aperture, ISA 10.2)
__device__ static inline unsigned lds_off(const void* p) {
  return (unsigned)(unsigned long long)p;
}
// Issue a 2-D TDM tile load: tile_h rows x tile_w elements (bf16, 2 bytes),
// global row stride row_stride elements, optional LDS padding bits
// (pad_en | pad_interval | pad_amount pre-packed for group1 word0).
__device__ static inline void tdm_load_2d(unsigned lds_byte,
                                          const void* gptr,
                                          unsigned tile_w, unsigned tile_h,
                                          unsigned long long row_stride,
                                          unsigned pad_bits) {
  unsigned long long ga = (unsigned long long)gptr;
  u32x4t g0 = { 1u,                 // count=1, user descriptor
                lds_byte,           // bits 63:32  lds_addr
                (unsigned)ga,       // bits 95:64  global_addr lo
                (unsigned)((ga >> 32) & 0x01FFFFFFu) | (2u << 30) }; // addr hi | type=2
  int w0 = (int)((1u << 16) | pad_bits);                    // data_size=2B + pad ctl
  int w1 = (int)((tile_w & 0xFFFFu) << 16);                 // tensor_dim0[15:0] @48
  int w2 = (int)(((tile_w >> 16) & 0xFFFFu) |               // tensor_dim0[31:16]
                 ((tile_h & 0xFFFFu) << 16));               // tensor_dim1[15:0]
  int w3 = (int)(((tile_h >> 16) & 0xFFFFu) |               // tensor_dim1[31:16]
                 ((tile_w & 0xFFFFu) << 16));               // tile_dim0
  int w4 = (int)(tile_h & 0xFFFFu);                         // tile_dim1 (tile_dim2=0)
  int w5 = (int)(row_stride & 0xFFFFFFFFull);               // dim0 stride lo32
  int w6 = (int)((row_stride >> 32) & 0xFFFFull);           // dim0 stride hi16
  int w7 = 0;
  i32x8t g1 = { w0, w1, w2, w3, w4, w5, w6, w7 };
  i32x4t gz4 = { 0, 0, 0, 0 };
  i32x8t gz8 = { 0, 0, 0, 0, 0, 0, 0, 0 };
  // clang-23 / amdgpu-toolchain 6-arg form:
  // (u32x4 g0, i32x8 g1, i32x4 g2, i32x4 g3, i32x8 pad, i32 cpol)
  __builtin_amdgcn_tensor_load_to_lds(g0, g1, gz4, gz4, gz8, 0);
}
// pad: enable | interval code 3 (16 DWORDs = 32 bf16) | amount code 3 (4 DWORDs = 8 bf16)
#define TDM_PAD_32x8 ((1u << 20) | (3u << 22) | (3u << 25))
#endif

// =====================================================================
// fp32 -> bf16 convert (grid-stride)
// =====================================================================
__global__ void cvt_f32_bf16_kernel(const float* __restrict__ in,
                                    bf16_t* __restrict__ out, long long n) {
  long long i = (long long)blockIdx.x * blockDim.x + threadIdx.x;
  long long st = (long long)gridDim.x * blockDim.x;
  for (; i < n; i += st) out[i] = (bf16_t)in[i];
}

// out_absorb transposed: wob_t[h][c][d] = Wkv_b[h][128+d][c]  (bf16)
__global__ void wobt_kernel(const float* __restrict__ w, bf16_t* __restrict__ out) {
  long long n = (long long)NHH * RANK * VDIM;
  long long i = (long long)blockIdx.x * blockDim.x + threadIdx.x;
  long long st = (long long)gridDim.x * blockDim.x;
  for (; i < n; i += st) {
    int h = (int)(i / (RANK * VDIM));
    int r = (int)(i % (RANK * VDIM));
    int c = r / VDIM;
    int d = r % VDIM;
    out[i] = (bf16_t)w[((long long)h * 256 + 128 + d) * RANK + c];
  }
}

// =====================================================================
// Generic bf16 GEMM: C[M,N] = A[M,K] * B[K,N], A/B row-major bf16,
// C fp32 or bf16.  Batched via blockIdx.z: z1 = z % zmod, z2 = z / zmod.
// BM=BN=128, BK=32, 256 threads = 8 waves (2 along M x 4 along N).
// A tile staged by the Tensor Data Mover (double buffered) when available.
// =====================================================================
template <bool OBF>
__global__ __launch_bounds__(256)
void gemm_bf16_kernel(const bf16_t* __restrict__ A, const bf16_t* __restrict__ Bm,
                      void* __restrict__ Cv,
                      int M, int N, int K, int lda, int ldb, int ldc,
                      long long aZ1, long long aZ2, long long bZ1, long long bZ2,
                      long long cZ1, long long cZ2, int zmod) {
#if HAS_TDM
  __shared__ bf16_t As[2][128][40];  // double buffered, TDM-filled, padded rows
#else
  __shared__ bf16_t As[1][128][40];
#endif
  __shared__ bf16_t Bs[128][40];     // [n][k] (transposed) padded

  int z = blockIdx.z;
  int z1 = z % zmod, z2 = z / zmod;
  const bf16_t* Ab = A + z1 * aZ1 + z2 * aZ2;
  const bf16_t* Bb = Bm + z1 * bZ1 + z2 * bZ2;

  int bm = blockIdx.y * 128, bn = blockIdx.x * 128;
  int tid = threadIdx.x, lane = tid & 31, wave = tid >> 5;
  int wm = (wave & 1) * 64;   // wave M origin inside tile
  int wn = (wave >> 1) * 32;  // wave N origin inside tile
  int r15 = lane & 15, hi = lane >> 4;
  bool fullN = (bn + 128 <= N);
  int nk = K >> 5;

  v8f acc[4][2];
#pragma unroll
  for (int mi = 0; mi < 4; ++mi)
#pragma unroll
    for (int ni = 0; ni < 2; ++ni) acc[mi][ni] = {};

#if HAS_TDM
  if (wave == 0)  // prologue: DMA tile 0 into buffer 0
    tdm_load_2d(lds_off(&As[0][0][0]), Ab + (long long)bm * lda, 32, 128,
                (unsigned long long)lda, TDM_PAD_32x8);
#endif

  for (int ki = 0; ki < nk; ++ki) {
    int k0 = ki << 5;
#if HAS_TDM
    int cur = ki & 1;
    if (wave == 0 && (ki + 1) < nk)  // DMA next tile into the other buffer
      tdm_load_2d(lds_off(&As[cur ^ 1][0][0]),
                  Ab + (long long)bm * lda + (k0 + 32), 32, 128,
                  (unsigned long long)lda, TDM_PAD_32x8);
#else
    int cur = 0;
    // ---- stage A tile synchronously: 512 u32x4, 2 per thread
#pragma unroll
    for (int l = 0; l < 2; ++l) {
      int it = tid + l * 256;
      int row = it >> 2, c16 = it & 3;
      *(u32x4*)(&As[0][row][c16 * 8]) =
          *(const u32x4*)(Ab + (long long)(bm + row) * lda + k0 + c16 * 8);
    }
#endif
    // ---- stage B tile transposed: 32 rows(k) x 128 cols(n)
#pragma unroll
    for (int l = 0; l < 2; ++l) {
      int it = tid + l * 256;
      int row = it >> 4;        // k index 0..31
      int c16 = it & 15;        // col group
      int n0 = c16 * 8;
      if (fullN) {
        u32x4 v = *(const u32x4*)(Bb + (long long)(k0 + row) * ldb + bn + n0);
        const bf16_t* pv = (const bf16_t*)&v;
#pragma unroll
        for (int i = 0; i < 8; ++i) Bs[n0 + i][row] = pv[i];
        if ((k0 + 32) < K)      // prefetch next B tile row chunk
          __builtin_prefetch(Bb + (long long)(k0 + 32 + row) * ldb + bn + n0, 0, 1);
      } else {
#pragma unroll
        for (int i = 0; i < 8; ++i) {
          int n = bn + n0 + i;
          Bs[n0 + i][row] =
              (n < N) ? Bb[(long long)(k0 + row) * ldb + n] : (bf16_t)0.0f;
        }
      }
    }
#if HAS_TDM
    if (wave == 0) {            // ensure tile `ki` has landed (next may be in flight)
      if ((ki + 1) < nk) __builtin_amdgcn_s_wait_tensorcnt(1);
      else               __builtin_amdgcn_s_wait_tensorcnt(0);
    }
#endif
    __syncthreads();

    // ---- fragments (ISA 16-bit layout: lane row, hi selects K-half pairs)
    FragB fa[4], fb[2];
#pragma unroll
    for (int mi = 0; mi < 4; ++mi) {
      int row = wm + mi * 16 + r15;
      fa[mi].u[0] = *(const u32x4*)(&As[cur][row][hi * 8]);
      fa[mi].u[1] = *(const u32x4*)(&As[cur][row][hi * 8 + 16]);
    }
#pragma unroll
    for (int ni = 0; ni < 2; ++ni) {
      int col = wn + ni * 16 + r15;
      fb[ni].u[0] = *(const u32x4*)(&Bs[col][hi * 8]);
      fb[ni].u[1] = *(const u32x4*)(&Bs[col][hi * 8 + 16]);
    }
#pragma unroll
    for (int mi = 0; mi < 4; ++mi)
#pragma unroll
      for (int ni = 0; ni < 2; ++ni)
        acc[mi][ni] = __builtin_amdgcn_wmma_f32_16x16x32_bf16(
            false, fa[mi].v, false, fb[ni].v, (short)0, acc[mi][ni], false, false);
    __syncthreads();
  }

  // ---- store (C 16x16 layout: VGPR r -> M=r (lanes 0-15) / M=r+8 (16-31))
  long long cOff = z1 * cZ1 + z2 * cZ2;
#pragma unroll
  for (int mi = 0; mi < 4; ++mi)
#pragma unroll
    for (int ni = 0; ni < 2; ++ni)
#pragma unroll
      for (int r = 0; r < 8; ++r) {
        int row = bm + wm + mi * 16 + r + hi * 8;
        int col = bn + wn + ni * 16 + r15;
        if (col < N) {
          float v = acc[mi][ni][r];
          if (OBF)
            ((bf16_t*)Cv)[cOff + (long long)row * ldc + col] = (bf16_t)v;
          else
            ((float*)Cv)[cOff + (long long)row * ldc + col] = v;
        }
      }
}

// =====================================================================
// prep_q: split q into q_nope (bf16) and RoPE'd q_pe -> qc[...,512:576]
// one block (128 threads) per (b,h,q) row
// =====================================================================
__global__ void prep_q_kernel(const bf16_t* __restrict__ qbf, const int* __restrict__ pos,
                              const float* __restrict__ cosc, const float* __restrict__ sinc,
                              bf16_t* __restrict__ qn, bf16_t* __restrict__ qc) {
  int id = blockIdx.x;                 // b*NH*S + h*S + q
  int b = id / (NHH * SS);
  int rem = id % (NHH * SS);
  int h = rem / SS;
  int q = rem % SS;
  int t = threadIdx.x;

  const bf16_t* src = qbf + ((long long)(b * SS + q)) * (NHH * QHD) + h * QHD;
  __shared__ float pe[ROPE];
  if (t < ROPE) pe[t] = (float)src[NOPE + t];
  // nope copy (bf16 -> bf16)
  qn[(((long long)(b * NHH + h) * SS) + q) * NOPE + t] = src[t];
  __syncthreads();
  if (t < ROPE) {
    int p = pos[b * SS + q];
    float c = cosc[(long long)p * ROPE + t];
    float s = sinc[(long long)p * ROPE + t];
    float o;
    if (t < 32) o = pe[2 * t] * c - pe[2 * t + 1] * s;          // d[j]*cos - d[j+32]*sin
    else { int j = t - 32; o = pe[2 * j + 1] * c + pe[2 * j] * s; }
    qc[(((long long)(b * NHH + h) * SS) + q) * DTOT + RANK + t] = (bf16_t)o;
  }
}

// =====================================================================
// prep_kv: RMS-norm compressed kv (512) + RoPE k_pe (64) -> kv_bf[b,k,576]
// one block (256 threads) per (b,k)
// =====================================================================
__global__ void prep_kv_kernel(const float* __restrict__ ckv, const int* __restrict__ pos,
                               const float* __restrict__ cosc, const float* __restrict__ sinc,
                               const float* __restrict__ lnw, bf16_t* __restrict__ kvb) {
  int id = blockIdx.x;  // b*S + k
  int t = threadIdx.x;
  const float* src = ckv + (long long)id * DTOT;
  __shared__ float red[256];
  __shared__ float pe[ROPE];
  __shared__ float rinv;

  float a0 = src[t], a1 = src[t + 256];
  red[t] = a0 * a0 + a1 * a1;
  if (t < ROPE) pe[t] = src[RANK + t];
  __syncthreads();
  for (int s = 128; s > 0; s >>= 1) {
    if (t < s) red[t] += red[t + s];
    __syncthreads();
  }
  if (t == 0) rinv = rsqrtf(red[0] / (float)RANK + 1e-6f);
  __syncthreads();

  bf16_t* dst = kvb + (long long)id * DTOT;
  dst[t]       = (bf16_t)(a0 * rinv * lnw[t]);
  dst[t + 256] = (bf16_t)(a1 * rinv * lnw[t + 256]);
  if (t < ROPE) {
    int p = pos[id];
    float c = cosc[(long long)p * ROPE + t];
    float s = sinc[(long long)p * ROPE + t];
    float o;
    if (t < 32) o = pe[2 * t] * c - pe[2 * t + 1] * s;
    else { int j = t - 32; o = pe[2 * j + 1] * c + pe[2 * j] * s; }
    dst[RANK + t] = (bf16_t)o;
  }
}

// =====================================================================
// Flash MLA attention.
//   Q tile: 32 q-rows x 576 (TDM-loaded; pitch 576, contiguous)
//   K tile: 32 keys  x 576 (pitch 584); V^T scattered at load
//   scores: split-K WMMA across wave pairs, online softmax, PV via WMMA
//   grid: (S/32, NH, B), 256 threads, ~124KB dynamic LDS
// =====================================================================
__global__ __launch_bounds__(256)
void mla_attn_kernel(const bf16_t* __restrict__ qc, const bf16_t* __restrict__ kv,
                     bf16_t* __restrict__ xb, float scale) {
  extern __shared__ char smem[];
  bf16_t* Qs = (bf16_t*)smem;                    // [32][576]  (no pad; TDM tile)
  bf16_t* Ks = (bf16_t*)(smem + 36864);          // [32][584]
  bf16_t* Vt = (bf16_t*)(smem + 74240);          // [512][40]  V transposed
  float*  Sp = (float*)(smem + 115200);          // [2][32][36] split-K partials
  bf16_t* Ps = (bf16_t*)(smem + 124416);         // [32][40]
  float*  mrow = (float*)(smem + 126976);        // [32]
  float*  lrow = mrow + 32;                      // [32]
  float*  arow = lrow + 32;                      // [32]

  int qt = blockIdx.x, h = blockIdx.y, b = blockIdx.z;
  int q0 = qt * 32;
  int tid = threadIdx.x, lane = tid & 31, wave = tid >> 5;
  int r15 = lane & 15, hi = lane >> 4;
  int wn0 = wave * 64;  // this wave owns output columns [wn0, wn0+64)

  const bf16_t* Qg = qc + (((long long)(b * NHH + h)) * SS + q0) * DTOT;
  const bf16_t* Kg = kv + ((long long)b * SS) * DTOT;

  // ---- load Q tile (32 x 576, fully contiguous)
#if HAS_TDM
  if (wave == 0)
    tdm_load_2d(lds_off(Qs), Qg, DTOT, 32, (unsigned long long)DTOT, 0);
#else
  for (int it = tid; it < (32 * 72); it += 256)
    ((u32x4*)Qs)[it] = ((const u32x4*)Qg)[it];
#endif
  if (tid < 32) { mrow[tid] = -3.0e38f; lrow[tid] = 0.0f; }

  v8f acc[2][4];
#pragma unroll
  for (int mi = 0; mi < 2; ++mi)
#pragma unroll
    for (int nc = 0; nc < 4; ++nc) acc[mi][nc] = {};
#if HAS_TDM
  if (wave == 0) __builtin_amdgcn_s_wait_tensorcnt(0);
#endif
  __syncthreads();

  for (int kt = 0; kt <= qt; ++kt) {
    int k0 = kt * 32;
    // ---- load K tile; also scatter V^T for the PV stage
    for (int it = tid; it < 32 * 72; it += 256) {
      int r = it / 72, c16 = it % 72;
      u32x4 v = *(const u32x4*)(Kg + (long long)(k0 + r) * DTOT + c16 * 8);
      *(u32x4*)(Ks + r * 584 + c16 * 8) = v;
      if (c16 < 64) {  // first 512 dims are V
        const bf16_t* pv = (const bf16_t*)&v;
        int d0 = c16 * 8;
#pragma unroll
        for (int i = 0; i < 8; ++i) Vt[(d0 + i) * 40 + r] = pv[i];
      }
      if (kt < qt)  // prefetch next K tile
        __builtin_prefetch(Kg + (long long)(k0 + 32 + r) * DTOT + c16 * 8, 0, 1);
    }
    __syncthreads();

    // ---- scores: 4 sub-tiles x 2 K-planes (288 each) across 8 waves
    {
      int sub = wave & 3, Mi = sub & 1, Ni = sub >> 1, plane = wave >> 2;
      int qrow = Mi * 16 + r15;
      int kcol = Ni * 16 + r15;
      v8f sc = {};
#pragma unroll
      for (int kk = 0; kk < 9; ++kk) {
        int d0 = plane * 288 + kk * 32;
        FragB fa, fb;
        fa.u[0] = *(const u32x4*)(Qs + qrow * 576 + d0 + hi * 8);
        fa.u[1] = *(const u32x4*)(Qs + qrow * 576 + d0 + hi * 8 + 16);
        fb.u[0] = *(const u32x4*)(Ks + kcol * 584 + d0 + hi * 8);
        fb.u[1] = *(const u32x4*)(Ks + kcol * 584 + d0 + hi * 8 + 16);
        sc = __builtin_amdgcn_wmma_f32_16x16x32_bf16(false, fa.v, false, fb.v,
                                                     (short)0, sc, false, false);
      }
      float* S0 = Sp + plane * (32 * 36);
#pragma unroll
      for (int r = 0; r < 8; ++r) {
        int row = Mi * 16 + r + hi * 8;
        S0[row * 36 + Ni * 16 + r15] = sc[r];
      }
    }
    __syncthreads();

    // ---- online softmax (row-parallel)
    if (tid < 32) {
      int row = tid, q = q0 + row;
      float mo = mrow[row];
      float mx = mo;
      float sv[32];
#pragma unroll
      for (int j = 0; j < 32; ++j) {
        float s = (Sp[row * 36 + j] + Sp[32 * 36 + row * 36 + j]) * scale;
        if (k0 + j > q) s = -1.0e9f;   // causal mask
        sv[j] = s;
        mx = fmaxf(mx, s);
      }
      float alpha = __expf(mo - mx);
      float sum = 0.0f;
#pragma unroll
      for (int j = 0; j < 32; ++j) {
        float p = __expf(sv[j] - mx);
        sum += p;
        Ps[row * 40 + j] = (bf16_t)p;
      }
      mrow[row] = mx;
      lrow[row] = lrow[row] * alpha + sum;
      arow[row] = alpha;
    }
    __syncthreads();

    // ---- rescale accumulators and P*V
    {
#pragma unroll
      for (int Mi = 0; Mi < 2; ++Mi)
#pragma unroll
        for (int r = 0; r < 8; ++r) {
          float a = arow[Mi * 16 + r + hi * 8];
#pragma unroll
          for (int nc = 0; nc < 4; ++nc) acc[Mi][nc][r] *= a;
        }
#pragma unroll
      for (int Mi = 0; Mi < 2; ++Mi) {
        FragB fa;
        int prow = Mi * 16 + r15;
        fa.u[0] = *(const u32x4*)(Ps + prow * 40 + hi * 8);
        fa.u[1] = *(const u32x4*)(Ps + prow * 40 + hi * 8 + 16);
#pragma unroll
        for (int nc = 0; nc < 4; ++nc) {
          int c = wn0 + nc * 16 + r15;
          FragB fb;
          fb.u[0] = *(const u32x4*)(Vt + c * 40 + hi * 8);
          fb.u[1] = *(const u32x4*)(Vt + c * 40 + hi * 8 + 16);
          acc[Mi][nc] = __builtin_amdgcn_wmma_f32_16x16x32_bf16(
              false, fa.v, false, fb.v, (short)0, acc[Mi][nc], false, false);
        }
      }
    }
    __syncthreads();
  }

  // ---- epilogue: divide by l, store bf16
  bf16_t* Og = xb + (((long long)(b * NHH + h)) * SS + q0) * RANK;
#pragma unroll
  for (int Mi = 0; Mi < 2; ++Mi)
#pragma unroll
    for (int nc = 0; nc < 4; ++nc)
#pragma unroll
      for (int r = 0; r < 8; ++r) {
        int row = Mi * 16 + r + hi * 8;
        float li = lrow[row];
        float v = acc[Mi][nc][r] / li;
        Og[(long long)row * RANK + wn0 + nc * 16 + r15] = (bf16_t)v;
      }
}

// =====================================================================
// host launcher
// =====================================================================
extern "C" void kernel_launch(void* const* d_in, const int* in_sizes, int n_in,
                              void* d_out, int out_size, void* d_ws, size_t ws_size,
                              hipStream_t stream) {
  (void)in_sizes; (void)n_in; (void)out_size; (void)ws_size;

  const float* hidden = (const float*)d_in[0];
  const int*   pos    = (const int*)d_in[2];
  const float* cosc   = (const float*)d_in[3];
  const float* sinc   = (const float*)d_in[4];
  const float* Wq     = (const float*)d_in[5];
  const float* Wkva   = (const float*)d_in[6];
  const float* lnw    = (const float*)d_in[7];
  const float* Wkvb   = (const float*)d_in[8];
  const float* Wo     = (const float*)d_in[9];

  char* ws = (char*)d_ws;
  size_t off = 0;
  auto carve = [&](size_t bytes) -> char* {
    char* p = ws + off;
    off = (off + bytes + 255) & ~(size_t)255;
    return p;
  };

  bf16_t* hbf     = (bf16_t*)carve((size_t)BB * SS * HH * 2);           // hidden bf16
  bf16_t* wq_bf   = (bf16_t*)carve((size_t)HH * NHH * QHD * 2);         // Wq
  bf16_t* wkva_bf = (bf16_t*)carve((size_t)HH * DTOT * 2);              // Wkv_a
  bf16_t* wkvb_bf = (bf16_t*)carve((size_t)NHH * 256 * RANK * 2);       // Wkv_b
  bf16_t* wob_t   = (bf16_t*)carve((size_t)NHH * RANK * VDIM * 2);      // out_absorb^T
  bf16_t* wo_bf   = (bf16_t*)carve((size_t)NHH * VDIM * HH * 2);        // Wo
  bf16_t* q_bf    = (bf16_t*)carve((size_t)BB * SS * NHH * QHD * 2);    // q proj
  float*  ckv_f32 = (float*)carve((size_t)BB * SS * DTOT * 4);          // ckv proj
  bf16_t* qn_bf   = (bf16_t*)carve((size_t)BB * NHH * SS * NOPE * 2);   // q_nope
  bf16_t* qc_bf   = (bf16_t*)carve((size_t)BB * NHH * SS * DTOT * 2);   // [q_nope_c|q_pe]
  bf16_t* kv_bf   = (bf16_t*)carve((size_t)BB * SS * DTOT * 2);         // [ckv|k_pe]
  bf16_t* xb_bf   = (bf16_t*)carve((size_t)BB * NHH * SS * RANK * 2);   // attn latent
  bf16_t* abf     = (bf16_t*)carve((size_t)BB * SS * NHH * VDIM * 2);   // attn out

  // ---- convert weights/activations to bf16
  cvt_f32_bf16_kernel<<<2048, 256, 0, stream>>>(hidden, hbf, (long long)BB * SS * HH);
  cvt_f32_bf16_kernel<<<2048, 256, 0, stream>>>(Wq, wq_bf, (long long)HH * NHH * QHD);
  cvt_f32_bf16_kernel<<<1024, 256, 0, stream>>>(Wkva, wkva_bf, (long long)HH * DTOT);
  cvt_f32_bf16_kernel<<<1024, 256, 0, stream>>>(Wkvb, wkvb_bf, (long long)NHH * 256 * RANK);
  cvt_f32_bf16_kernel<<<2048, 256, 0, stream>>>(Wo, wo_bf, (long long)NHH * VDIM * HH);
  wobt_kernel<<<1024, 256, 0, stream>>>(Wkvb, wob_t);

  // ---- G1: q = hidden @ Wq  (4096 x 3072 x 2048), bf16 out
  {
    dim3 g(NHH * QHD / 128, BB * SS / 128, 1);
    gemm_bf16_kernel<true><<<g, 256, 0, stream>>>(
        hbf, wq_bf, q_bf, BB * SS, NHH * QHD, HH, HH, NHH * QHD, NHH * QHD,
        0, 0, 0, 0, 0, 0, 1);
  }
  prep_q_kernel<<<BB * NHH * SS, 128, 0, stream>>>(q_bf, pos, cosc, sinc, qn_bf, qc_bf);

  // ---- G2: ckv = hidden @ Wkv_a  (4096 x 576 x 2048), f32 out
  {
    dim3 g((DTOT + 127) / 128, BB * SS / 128, 1);
    gemm_bf16_kernel<false><<<g, 256, 0, stream>>>(
        hbf, wkva_bf, ckv_f32, BB * SS, DTOT, HH, HH, DTOT, DTOT,
        0, 0, 0, 0, 0, 0, 1);
  }
  prep_kv_kernel<<<BB * SS, 256, 0, stream>>>(ckv_f32, pos, cosc, sinc, lnw, kv_bf);

  // ---- G3: per-(b,h) absorb: qc[...,0:512] = qn (S x 128) @ q_absorb (128 x 512)
  {
    dim3 g(RANK / 128, SS / 128, BB * NHH);
    gemm_bf16_kernel<true><<<g, 256, 0, stream>>>(
        qn_bf, wkvb_bf, qc_bf, SS, RANK, NOPE, NOPE, RANK, DTOT,
        (long long)SS * NOPE, (long long)NHH * SS * NOPE,
        (long long)256 * RANK, 0,
        (long long)SS * DTOT, (long long)NHH * SS * DTOT, NHH);
  }

  // ---- flash attention
  {
    dim3 g(SS / 32, NHH, BB);
    float scale = 1.0f / sqrtf((float)QHD);
    mla_attn_kernel<<<g, 256, 127360, stream>>>(qc_bf, kv_bf, xb_bf, scale);
  }

  // ---- G5: per-(b,h) out-absorb: attn = x (S x 512) @ out_absorb^T (512 x 128)
  {
    dim3 g(VDIM / 128, SS / 128, BB * NHH);
    gemm_bf16_kernel<true><<<g, 256, 0, stream>>>(
        xb_bf, wob_t, abf, SS, VDIM, RANK, RANK, VDIM, NHH * VDIM,
        (long long)SS * RANK, (long long)NHH * SS * RANK,
        (long long)RANK * VDIM, 0,
        (long long)VDIM, (long long)SS * NHH * VDIM, NHH);
  }

  // ---- G6: out = attn_flat (4096 x 2048) @ Wo (2048 x 2048), f32 out
  {
    dim3 g(HH / 128, BB * SS / 128, 1);
    gemm_bf16_kernel<false><<<g, 256, 0, stream>>>(
        abf, wo_bf, (float*)d_out, BB * SS, HH, NHH * VDIM,
        NHH * VDIM, HH, HH, 0, 0, 0, 0, 0, 0, 1);
  }
}
